// Attention_2482491097466
// MI455X (gfx1250) — matrix operand order repaired and checked by reading, not compiled, not run on hardware
//
#include <hip/hip_runtime.h>
#include <math.h>

// Problem constants (from reference): L=1024, N=8, C=1024, H=16, D=64
#define L_SEQ   1024
#define NBATCH  8
#define CDIM    1024
#define HHEADS  16
#define DHEAD   64
#define NHB     (NBATCH * HHEADS)   // 128 head-batches
#define C3      (3 * CDIM)          // 3072

typedef __bf16 bf16_t;
typedef __attribute__((ext_vector_type(16))) __bf16 bf16x16;
typedef __attribute__((ext_vector_type(8)))  float  f32x8;

// ---- bf16 <-> f32 helpers (bit-exact RNE, no reliance on __bf16 conversions) ----
__device__ __forceinline__ bf16_t f2bf(float f) {
  union { float f; unsigned u; } v; v.f = f;
  unsigned r = (v.u + 0x7FFFu + ((v.u >> 16) & 1u)) >> 16;
  union { unsigned short s; bf16_t b; } o; o.s = (unsigned short)r; return o.b;
}

__device__ __forceinline__ f32x8 wmma_bf16(bf16x16 a, bf16x16 b, f32x8 c) {
  return __builtin_amdgcn_wmma_f32_16x16x32_bf16(false, a, false, b, (short)0, c,
                                                 false, false);
}

// ---- CDNA5 async global->LDS DMA (ASYNCcnt-tracked), inline asm for
//      toolchain portability (builtin arity differs between toolchains) ----
__device__ __forceinline__ unsigned lds_off(const void* p) {
  // LDS aperture: low 32 bits of the generic address are the LDS byte address
  return (unsigned)(size_t)p;
}
__device__ __forceinline__ void async_b128(unsigned lds_addr, const void* gaddr) {
  asm volatile("global_load_async_to_lds_b128 %0, %1, off"
               :: "v"(lds_addr), "v"(gaddr) : "memory");
}
__device__ __forceinline__ void async_wait0() {
  asm volatile("s_wait_asynccnt 0x0" ::: "memory");
}

// =====================================================================
// Kernel 0: f32 -> bf16 cast (x, w_in, w_out)
// =====================================================================
__global__ void cast_f32_bf16(const float* __restrict__ src,
                              bf16_t* __restrict__ dst, int n) {
  int i = blockIdx.x * blockDim.x + threadIdx.x;
  if (i < n) dst[i] = f2bf(src[i]);
}

// =====================================================================
// Kernel 1: QKV projection + bias + L2-normalize(q,k) + head_scale*v,
//           scattered to (NH, L, D) bf16.
// grid: (C3/64, L*N/16), block: 128 threads (4 waves).
// A-tile double-buffered via async DMA into LDS.
// =====================================================================
__global__ void qkv_kernel(const bf16_t* __restrict__ xh,     // (L*N, C)
                           const bf16_t* __restrict__ w_in_h, // (3C, C)
                           const float*  __restrict__ b_in,   // (3C)
                           const float*  __restrict__ head_scale, // (H)
                           bf16_t* __restrict__ qn,           // (NH, L, D)
                           bf16_t* __restrict__ kn,           // (NH, L, D)
                           bf16_t* __restrict__ vh)           // (NH, L, D)
{
  __shared__ bf16_t sA[2][16 * 32]; // double-buffered A k-tile (1 KB each)
  __shared__ float  sOut[16 * 64];  // full f32 output tile
  __shared__ float  sNorm[16];

  const int tid   = threadIdx.x;
  const int wave  = tid >> 5;
  const int lane  = tid & 31;
  const int ncol  = lane & 15;
  const int khalf = lane >> 4;

  const int tile_n = blockIdx.x;        // 0..47
  const int row0   = blockIdx.y * 16;   // row tile in (L*N)
  const int col0   = tile_n * 64 + wave * 16;

  // stage: 16x32 bf16 = 64 x 16B chunks, one async b128 per thread (tid<64)
  auto stage = [&](int buf, int k0) {
    if (tid < 64) {
      const int r = tid >> 2, ch = tid & 3;
      async_b128(lds_off(&sA[buf][0]) + tid * 16,
                 &xh[(size_t)(row0 + r) * CDIM + k0 + ch * 8]);
    }
  };

  f32x8 acc = {0.f, 0.f, 0.f, 0.f, 0.f, 0.f, 0.f, 0.f};

  int p = 0;
  stage(0, 0);
  for (int k0 = 0; k0 < CDIM; k0 += 32) {
    async_wait0();        // own async stores into sA[p] complete
    __syncthreads();      // everyone's complete; prior reads of sA[p^1] drained
    if (k0 + 32 < CDIM) {
      stage(p ^ 1, k0 + 32);
      __builtin_prefetch(&w_in_h[(size_t)(col0 + ncol) * CDIM + k0 + 32], 0, 1);
    }

    // A fragment (16x32 bf16): lane holds row (lane&15); K pairs per VGPR
    bf16x16 a;
    {
      const bf16_t* pa = &sA[p][(lane & 15) * 32 + khalf * 8];
#pragma unroll
      for (int j = 0; j < 8; ++j) { a[j] = pa[j]; a[8 + j] = pa[16 + j]; }
    }
    // B fragment (32x16): B(k,n) = w_in[n,k]; two 8-bf16 runs, k = 8h+j and 16+8h+j, as for A
    bf16x16 b;
    {
      const bf16_t* pb = &w_in_h[(size_t)(col0 + ncol) * CDIM + k0 + khalf * 8];
#pragma unroll
      for (int j = 0; j < 8; ++j) { b[j] = pb[j]; b[8 + j] = pb[16 + j]; }
    }
    acc = wmma_bf16(a, b, acc);
    p ^= 1;
  }

  // bias + dump f32 tile to LDS
  const float bias = b_in[col0 + ncol];
  __syncthreads();
#pragma unroll
  for (int r = 0; r < 8; ++r) {
    sOut[(r + khalf * 8) * 64 + wave * 16 + ncol] = acc[r] + bias;
  }
  __syncthreads();

  const int sec = tile_n / 16;          // 0=q, 1=k, 2=v
  const int h   = tile_n % 16;

  if (sec < 2) {  // row L2 norms over D=64 (uniform branch per block)
    if (tid < 16) {
      float s = 0.f;
#pragma unroll
      for (int d = 0; d < 64; ++d) { float t = sOut[tid * 64 + d]; s += t * t; }
      sNorm[tid] = fmaxf(sqrtf(s), 1e-12f);
    }
    __syncthreads();
  }

  bf16_t* dst = (sec == 0) ? qn : ((sec == 1) ? kn : vh);
  const float hs = (sec == 2) ? head_scale[h] : 1.0f;
  for (int e = tid; e < 16 * 64; e += 128) {
    const int r = e >> 6, d = e & 63;
    const int m = row0 + r;
    const int l = m / NBATCH, n = m % NBATCH;
    const int bh = n * HHEADS + h;
    float v = sOut[e];
    v = (sec < 2) ? (v / sNorm[r]) : (v * hs);
    dst[((size_t)bh * L_SEQ + l) * DHEAD + d] = f2bf(v);
  }
}

// =====================================================================
// Kernel 2: logits = qn . kn^T, * exp(min(logit_scale,log100)), softmax,
//           write f32 attn. grid: (NH, L/16), block 256 (8 waves).
// Dynamic LDS: 16*1024 f32 logits + 16*16 f32 reduce scratch.
// =====================================================================
__global__ void attn_kernel(const bf16_t* __restrict__ qn,
                            const bf16_t* __restrict__ kn,
                            const float*  __restrict__ logit_scale, // (H)
                            float* __restrict__ attn)               // (NH,L,L)
{
  extern __shared__ float sm[];
  float* slog = sm;                 // [16][1024]
  float* sred = sm + 16 * 1024;     // [16][16]

  const int b  = blockIdx.x;        // head-batch = n*H + h
  const int r0 = blockIdx.y * 16;
  const int h  = b % HHEADS;

  const int tid   = threadIdx.x;
  const int wave  = tid >> 5;
  const int lane  = tid & 31;
  const int ncol  = lane & 15;
  const int khalf = lane >> 4;

  const float LOG100 = 4.605170185988091f;
  const float ls = __expf(fminf(logit_scale[h], LOG100));

  // A fragments: 16 q-rows x D=64 -> two K=32 fragments, loaded once
  bf16x16 a0, a1;
  {
    const bf16_t* qrow = qn + ((size_t)b * L_SEQ + r0 + (lane & 15)) * DHEAD;
#pragma unroll
    for (int j = 0; j < 8; ++j) {
      a0[j]     = qrow[khalf * 8 + j];
      a0[8 + j] = qrow[16 + khalf * 8 + j];
      a1[j]     = qrow[32 + khalf * 8 + j];
      a1[8 + j] = qrow[48 + khalf * 8 + j];
    }
  }

  // 64 column tiles of 16; wave w handles t = w, w+8, ...
  for (int t = wave; t < 64; t += 8) {
    const int c0 = t * 16;
    f32x8 acc = {0.f, 0.f, 0.f, 0.f, 0.f, 0.f, 0.f, 0.f};
    const bf16_t* krow = kn + ((size_t)b * L_SEQ + c0 + ncol) * DHEAD;
    __builtin_prefetch(krow + 128 * DHEAD, 0, 1); // next tile for this wave
    bf16x16 b0, b1;
#pragma unroll
    for (int j = 0; j < 8; ++j) {
      b0[j]     = krow[khalf * 8 + j];
      b0[8 + j] = krow[16 + khalf * 8 + j];
      b1[j]     = krow[32 + khalf * 8 + j];
      b1[8 + j] = krow[48 + khalf * 8 + j];
    }
    acc = wmma_bf16(a0, b0, acc);
    acc = wmma_bf16(a1, b1, acc);
#pragma unroll
    for (int r = 0; r < 8; ++r)
      slog[(r + khalf * 8) * 1024 + c0 + ncol] = acc[r] * ls;
  }
  __syncthreads();

  // softmax: 16 threads per row, each owning 4 contiguous cols per 64-col group
  const int r  = tid >> 4;
  const int c0 = (tid & 15) * 4;

  float m = -INFINITY;
#pragma unroll
  for (int i = 0; i < 16; ++i) {
    const float4 v = *(const float4*)&slog[r * 1024 + i * 64 + c0];
    m = fmaxf(m, fmaxf(fmaxf(v.x, v.y), fmaxf(v.z, v.w)));
  }
  sred[r * 16 + (tid & 15)] = m;
  __syncthreads();
  if ((tid & 15) == 0) {
    float mm = sred[r * 16];
#pragma unroll
    for (int j = 1; j < 16; ++j) mm = fmaxf(mm, sred[r * 16 + j]);
    sred[r * 16] = mm;
  }
  __syncthreads();
  const float rowmax = sred[r * 16];
  __syncthreads();

  float s = 0.f;
#pragma unroll
  for (int i = 0; i < 16; ++i) {
    float4* pp = (float4*)&slog[r * 1024 + i * 64 + c0];
    float4 v = *pp;
    v.x = __expf(v.x - rowmax);
    v.y = __expf(v.y - rowmax);
    v.z = __expf(v.z - rowmax);
    v.w = __expf(v.w - rowmax);
    s += (v.x + v.y) + (v.z + v.w);
    *pp = v;
  }
  sred[r * 16 + (tid & 15)] = s;
  __syncthreads();
  if ((tid & 15) == 0) {
    float ss = 0.f;
#pragma unroll
    for (int j = 0; j < 16; ++j) ss += sred[r * 16 + j];
    sred[r * 16] = ss;
  }
  __syncthreads();
  const float inv = 1.0f / sred[r * 16];

  float* dst = attn + ((size_t)b * L_SEQ + r0 + r) * L_SEQ;
#pragma unroll
  for (int i = 0; i < 16; ++i) {
    float4 v = *(const float4*)&slog[r * 1024 + i * 64 + c0];
    v.x *= inv; v.y *= inv; v.z *= inv; v.w *= inv;
    *(float4*)&dst[i * 64 + c0] = v;          // coalesced b128 stores
  }
}

// =====================================================================
// Kernel 3: out_h = attn @ v (head_scale already folded into v),
//           merged back to (L*N, C) bf16. grid: (NH, L/16), block 128.
// V-tile double-buffered via async DMA into LDS.
// =====================================================================
__global__ void av_kernel(const float*  __restrict__ attn, // (NH, L, L) f32
                          const bf16_t* __restrict__ vh,   // (NH, L, D)
                          bf16_t* __restrict__ attout)     // (L*N, C)
{
  __shared__ bf16_t sV[2][32 * 64]; // double-buffered v tile (4 KB each)

  const int b  = blockIdx.x;
  const int r0 = blockIdx.y * 16;
  const int h  = b % HHEADS;
  const int n  = b / HHEADS;

  const int tid   = threadIdx.x;
  const int wave  = tid >> 5;
  const int lane  = tid & 31;
  const int ncol  = lane & 15;
  const int khalf = lane >> 4;

  // stage: 32x64 bf16 = 4 KB = 256 x 16B chunks; 2 async b128 per thread
  auto stageV = [&](int buf, int k0) {
    for (int i = tid; i < 256; i += 128) {
      const int r = i >> 3, ch = i & 7;
      async_b128(lds_off(&sV[buf][0]) + i * 16,
                 &vh[((size_t)b * L_SEQ + k0 + r) * DHEAD + ch * 8]);
    }
  };

  f32x8 acc = {0.f, 0.f, 0.f, 0.f, 0.f, 0.f, 0.f, 0.f};
  const float* arow = attn + ((size_t)b * L_SEQ + r0 + (lane & 15)) * L_SEQ;

  int p = 0;
  stageV(0, 0);
  for (int k0 = 0; k0 < L_SEQ; k0 += 32) {
    async_wait0();
    __syncthreads();
    if (k0 + 32 < L_SEQ) {
      stageV(p ^ 1, k0 + 32);
      __builtin_prefetch(&arow[k0 + 32], 0, 1);
    }

    // A fragment: softmax weights f32 -> bf16 on the fly
    bf16x16 a;
#pragma unroll
    for (int j = 0; j < 8; ++j) {
      a[j]     = f2bf(arow[k0 + khalf * 8 + j]);
      a[8 + j] = f2bf(arow[k0 + 16 + khalf * 8 + j]);
    }
    // B fragment from LDS: B(k, d) = v[k, wave*16 + ncol]
    bf16x16 bb;
#pragma unroll
    for (int j = 0; j < 16; ++j)
      bb[j] = sV[p][((j >> 3) * 16 + khalf * 8 + (j & 7)) * 64 + wave * 16 + ncol];
    acc = wmma_bf16(a, bb, acc);
    p ^= 1;
  }

  const int c = h * DHEAD + wave * 16 + ncol;
#pragma unroll
  for (int r = 0; r < 8; ++r) {
    const int l = r0 + r + khalf * 8;
    attout[((size_t)l * NBATCH + n) * CDIM + c] = f2bf(acc[r]);
  }
}

// =====================================================================
// Kernel 4: out = attout @ w_out^T + b_out (f32 result).
// grid: (C/64, L*N/16), block 128 (4 waves). uint4 LDS staging.
// =====================================================================
__global__ void proj_kernel(const bf16_t* __restrict__ attout,  // (L*N, C)
                            const bf16_t* __restrict__ w_out_h, // (C, C)
                            const float*  __restrict__ b_out,   // (C)
                            float* __restrict__ out)            // (L*N, C)
{
  __shared__ bf16_t sA[16 * 32];

  const int tid   = threadIdx.x;
  const int wave  = tid >> 5;
  const int lane  = tid & 31;
  const int ncol  = lane & 15;
  const int khalf = lane >> 4;

  const int row0 = blockIdx.y * 16;
  const int col0 = blockIdx.x * 64 + wave * 16;

  f32x8 acc = {0.f, 0.f, 0.f, 0.f, 0.f, 0.f, 0.f, 0.f};

  for (int k0 = 0; k0 < CDIM; k0 += 32) {
    __syncthreads();
    if (tid < 64) {  // 64 x 16B vector copies: global_load_b128 + ds_store_b128
      const int r = tid >> 2, ch = tid & 3;
      ((uint4*)sA)[tid] =
          *(const uint4*)&attout[(size_t)(row0 + r) * CDIM + k0 + ch * 8];
    }
    if (k0 + 32 < CDIM)
      __builtin_prefetch(&w_out_h[(size_t)(col0 + ncol) * CDIM + k0 + 32], 0, 1);
    __syncthreads();

    bf16x16 a;
    {
      const bf16_t* pa = &sA[(lane & 15) * 32 + khalf * 8];
#pragma unroll
      for (int j = 0; j < 8; ++j) { a[j] = pa[j]; a[8 + j] = pa[16 + j]; }
    }
    bf16x16 bfr;
    {
      const bf16_t* pb = &w_out_h[(size_t)(col0 + ncol) * CDIM + k0 + khalf * 8];
#pragma unroll
      for (int j = 0; j < 8; ++j) { bfr[j] = pb[j]; bfr[8 + j] = pb[16 + j]; }
    }
    acc = wmma_bf16(a, bfr, acc);
  }

  const float bias = b_out[col0 + ncol];
#pragma unroll
  for (int r = 0; r < 8; ++r) {
    out[(size_t)(row0 + r + khalf * 8) * CDIM + col0 + ncol] = acc[r] + bias;
  }
}

// =====================================================================
// Host launcher
// =====================================================================
extern "C" void kernel_launch(void* const* d_in, const int* in_sizes, int n_in,
                              void* d_out, int out_size, void* d_ws, size_t ws_size,
                              hipStream_t stream) {
  const float* x           = (const float*)d_in[0];
  const float* w_in        = (const float*)d_in[1];
  const float* b_in        = (const float*)d_in[2];
  const float* logit_scale = (const float*)d_in[3];
  const float* head_scale  = (const float*)d_in[4];
  const float* w_out       = (const float*)d_in[5];
  const float* b_out       = (const float*)d_in[6];

  float* out  = (float*)d_out;
  float* attn = out + (size_t)L_SEQ * NBATCH * CDIM;  // outputs concatenated

  // workspace carve-up (bf16)
  char* ws = (char*)d_ws;
  bf16_t* xh      = (bf16_t*)ws; ws += (size_t)L_SEQ * NBATCH * CDIM * 2;
  bf16_t* w_in_h  = (bf16_t*)ws; ws += (size_t)C3 * CDIM * 2;
  bf16_t* w_out_h = (bf16_t*)ws; ws += (size_t)CDIM * CDIM * 2;
  bf16_t* qn      = (bf16_t*)ws; ws += (size_t)NHB * L_SEQ * DHEAD * 2;
  bf16_t* kn      = (bf16_t*)ws; ws += (size_t)NHB * L_SEQ * DHEAD * 2;
  bf16_t* vh      = (bf16_t*)ws; ws += (size_t)NHB * L_SEQ * DHEAD * 2;
  bf16_t* attout  = (bf16_t*)ws; ws += (size_t)L_SEQ * NBATCH * CDIM * 2;

  // casts
  {
    int n0 = L_SEQ * NBATCH * CDIM;
    cast_f32_bf16<<<(n0 + 255) / 256, 256, 0, stream>>>(x, xh, n0);
    int n1 = C3 * CDIM;
    cast_f32_bf16<<<(n1 + 255) / 256, 256, 0, stream>>>(w_in, w_in_h, n1);
    int n2 = CDIM * CDIM;
    cast_f32_bf16<<<(n2 + 255) / 256, 256, 0, stream>>>(w_out, w_out_h, n2);
  }

  // QKV + normalize + head scatter
  {
    dim3 g(C3 / 64, (L_SEQ * NBATCH) / 16);
    qkv_kernel<<<g, 128, 0, stream>>>(xh, w_in_h, b_in, head_scale, qn, kn, vh);
  }
  // logits + softmax -> attn (f32, second output)
  {
    dim3 g(NHB, L_SEQ / 16);
    size_t shmem = (16 * 1024 + 16 * 16) * sizeof(float);
    attn_kernel<<<g, 256, shmem, stream>>>(qn, kn, logit_scale, attn);
  }
  // attn @ v -> merged heads (bf16)
  {
    dim3 g(NHB, L_SEQ / 16);
    av_kernel<<<g, 128, 0, stream>>>(attn, vh, attout);
  }
  // output projection -> out (f32, first output)
  {
    dim3 g(CDIM / 64, (L_SEQ * NBATCH) / 16);
    proj_kernel<<<g, 128, 0, stream>>>(attout, w_out_h, b_out, out);
  }
}
